// Attention_52132313039212
// MI455X (gfx1250) — matrix operand order repaired and checked by reading, not compiled, not run on hardware
//
#include <hip/hip_runtime.h>
#include <hip/hip_bf16.h>

typedef __attribute__((ext_vector_type(16))) _Float16 v16h;
typedef __attribute__((ext_vector_type(8)))  _Float16 v8h;
typedef __attribute__((ext_vector_type(8)))  float    v8f;

#define S_LEN 2048
#define D_MODEL 2048
#define NHEADS 16
#define HDIM 128
#define DMH 256            // head dim * C (C=2)
#define QKN (NHEADS*DMH)   // 4096
#define INV_SQRT_HD 0.08838834764831845f
#define EPS_F 1e-9f

// ---------------------------------------------------------------------------
// WMMA helper: D = A(16x32 f16) * B(32x16 f16) + C(16x16 f32)
// ---------------------------------------------------------------------------
__device__ __forceinline__ v8f wmma_f16(v16h a, v16h b, v8f c) {
    return __builtin_amdgcn_wmma_f32_16x16x32_f16(false, a, false, b,
                                                  (short)0, c, false, false);
}

// A-fragment (16x32, M x K) from row-major memory, lda in elements.
// ISA layout: lane m (lo half): K {0..7, 16..23}; lane m+16: K {8..15, 24..31}.
__device__ __forceinline__ v16h load_a_frag(const _Float16* A, int lda,
                                            int row0, int k0, int lane) {
    int half = lane >> 4, m = lane & 15;
    const _Float16* p = A + (size_t)(row0 + m) * lda + k0 + half * 8;
    v8h lo = *(const v8h*)p;          // K +0..7   (or +8..15)
    v8h hi = *(const v8h*)(p + 16);   // K +16..23 (or +24..31)
    v16h a;
#pragma unroll
    for (int i = 0; i < 8; ++i) { a[i] = lo[i]; a[i + 8] = hi[i]; }
    return a;
}

// B-fragment (32x16, K x N) where B[k][n] = Bw[col0+n][k0+k] (row-major Bw).
// ISA layout: lane n (lo half): K {0..7, 16..23}; lane n+16: K {8..15, 24..31}.
__device__ __forceinline__ v16h load_b_frag(const _Float16* Bw, int ldb,
                                            int col0, int k0, int lane) {
    int half = lane >> 4, n = lane & 15;
    const _Float16* p = Bw + (size_t)(col0 + n) * ldb + k0 + half * 8;
    v8h lo = *(const v8h*)p;
    v8h hi = *(const v8h*)(p + 16);
    v16h b;
#pragma unroll
    for (int i = 0; i < 8; ++i) { b[i] = lo[i]; b[i + 8] = hi[i]; }
    return b;
}

// ---------------------------------------------------------------------------
// 1) Row RMSNorm: f32 -> f16
// ---------------------------------------------------------------------------
__global__ __launch_bounds__(256)
void rmsnorm_kernel(const float* __restrict__ x, const float* __restrict__ w,
                    _Float16* __restrict__ out, int D) {
    __shared__ float red[256];
    int row = blockIdx.x;
    const float* xr = x + (size_t)row * D;
    float ss = 0.f;
    for (int j = threadIdx.x; j < D; j += 256) ss += xr[j] * xr[j];
    red[threadIdx.x] = ss; __syncthreads();
    for (int s = 128; s > 0; s >>= 1) {
        if (threadIdx.x < s) red[threadIdx.x] += red[threadIdx.x + s];
        __syncthreads();
    }
    float r = rsqrtf(red[0] / (float)D + EPS_F);
    for (int j = threadIdx.x; j < D; j += 256)
        out[(size_t)row * D + j] = (_Float16)(xr[j] * r * w[j]);
}

// ---------------------------------------------------------------------------
// 2) f32 -> f16 elementwise convert (for weights)
// ---------------------------------------------------------------------------
__global__ __launch_bounds__(256)
void cvt_f32_f16_kernel(const float* __restrict__ in, _Float16* __restrict__ out,
                        size_t n) {
    size_t i = (size_t)blockIdx.x * 256 + threadIdx.x;
    if (i < n) out[i] = (_Float16)in[i];
}

// ---------------------------------------------------------------------------
// 3) WMMA GEMM: C[M,N] = A[M,K] * Bw[N,K]^T  (A,Bw f16 row-major)
//    8 waves/block, each wave computes a 32x64 tile -> block tile 128x128.
//    8 WMMAs per K-step per wave (0.67 wmma per b128 load) + speculative
//    global_prefetch of the K-tile 128 elements ahead.
//    If Cf != nullptr: Cf = acc + resid (f32).  Else: Ch = (f16)acc.
// ---------------------------------------------------------------------------
__global__ __launch_bounds__(256)
void gemm_f16_wmma_kernel(const _Float16* __restrict__ A,
                          const _Float16* __restrict__ Bw,
                          int M, int N, int K,
                          _Float16* __restrict__ Ch,
                          float* __restrict__ Cf,
                          const float* __restrict__ resid) {
    int lane = threadIdx.x & 31;
    int wave = threadIdx.x >> 5;          // 0..7
    int row0 = blockIdx.x * 128 + (wave & 3) * 32;
    int col0 = blockIdx.y * 128 + (wave >> 2) * 64;

    v8f acc[2][4] = {};
    for (int k0 = 0; k0 < K; k0 += 32) {
        v16h a0 = load_a_frag(A, K, row0,      k0, lane);
        v16h a1 = load_a_frag(A, K, row0 + 16, k0, lane);
        if (k0 + 128 < K) {
            // speculative stream-ahead into cache (global_prefetch_b8)
            __builtin_prefetch(A  + (size_t)(row0 + (lane & 15)) * K + k0 + 128, 0, 1);
            __builtin_prefetch(Bw + (size_t)(col0 + lane)        * K + k0 + 128, 0, 1);
            __builtin_prefetch(Bw + (size_t)(col0 + 32 + lane)   * K + k0 + 128, 0, 1);
        }
#pragma unroll
        for (int ni = 0; ni < 4; ++ni) {
            v16h b = load_b_frag(Bw, K, col0 + ni * 16, k0, lane);
            acc[0][ni] = wmma_f16(a0, b, acc[0][ni]);
            acc[1][ni] = wmma_f16(a1, b, acc[1][ni]);
        }
    }

    int half = lane >> 4, nn = lane & 15;
#pragma unroll
    for (int mi = 0; mi < 2; ++mi)
#pragma unroll
        for (int ni = 0; ni < 4; ++ni)
#pragma unroll
            for (int r = 0; r < 8; ++r) {
                int row = row0 + mi * 16 + r + half * 8;
                int col = col0 + ni * 16 + nn;
                size_t idx = (size_t)row * N + col;
                if (Cf) Cf[idx] = acc[mi][ni][r] + resid[idx];
                else    Ch[idx] = (_Float16)acc[mi][ni][r];
            }
}

// ---------------------------------------------------------------------------
// 4) RoPE in place on f16 tensor laid out [s][h*DMv + j]
// ---------------------------------------------------------------------------
__global__ __launch_bounds__(256)
void rope_kernel(_Float16* __restrict__ t, int nheads, int DMv) {
    int pairs = DMv / 2;
    int total = S_LEN * nheads * pairs;
    int idx = blockIdx.x * 256 + threadIdx.x;
    if (idx >= total) return;
    int i = idx % pairs; int tmp = idx / pairs;
    int h = tmp % nheads; int s = tmp / nheads;
    float theta = __powf(10000.0f, -(float)(2 * i) / (float)DMv);
    float sn, cs; __sincosf((float)s * theta, &sn, &cs);
    size_t base = ((size_t)s * nheads + h) * DMv + 2 * i;
    float xr = (float)t[base], xi = (float)t[base + 1];
    t[base]     = (_Float16)(xr * cs - xi * sn);
    t[base + 1] = (_Float16)(xr * sn + xi * cs);
}

// ---------------------------------------------------------------------------
// 5) V transpose: v[s][h*128+d] -> vT[h][d][s]  (f16)
// ---------------------------------------------------------------------------
__global__ __launch_bounds__(256)
void vtrans_kernel(const _Float16* __restrict__ v, _Float16* __restrict__ vT) {
    size_t idx = (size_t)blockIdx.x * 256 + threadIdx.x;   // S*2048
    int col = (int)(idx & 2047); int s = (int)(idx >> 11);
    int h = col >> 7, d = col & 127;
    vT[((size_t)h * HDIM + d) * S_LEN + s] = v[idx];
}

// ---------------------------------------------------------------------------
// 6) per-head differential scale
// ---------------------------------------------------------------------------
__global__ void scale_kernel(const float* __restrict__ q1, const float* __restrict__ q2,
                             const float* __restrict__ k1, const float* __restrict__ k2,
                             const float* __restrict__ lam, float* __restrict__ out) {
    int h = threadIdx.x;
    if (h >= NHEADS) return;
    float s1 = 0.f, s2 = 0.f;
    for (int d = 0; d < HDIM; ++d) {
        s1 += q1[h * HDIM + d] * k1[h * HDIM + d];
        s2 += q2[h * HDIM + d] * k2[h * HDIM + d];
    }
    float base = __expf(s1) - __expf(s2);
    float lsum = 0.f;
    for (int i = 0; i < NHEADS; ++i) lsum += lam[i];
    out[h] = -((float)NHEADS * base + lsum);
}

// ---------------------------------------------------------------------------
// 7) Flash-style differential attention.
//    grid = (S/16, H), block = 64 (2 waves).  Wave c handles channel c.
//    Online softmax, 32 keys/iteration, WMMA for QK^T and P*V.
// ---------------------------------------------------------------------------
__global__ __launch_bounds__(64)
void flashdiff_kernel(const _Float16* __restrict__ Q, const _Float16* __restrict__ Kk,
                      const _Float16* __restrict__ VT, const float* __restrict__ scaleArr,
                      const float* __restrict__ head_w, const float* __restrict__ lam,
                      _Float16* __restrict__ attOut) {
    __shared__ __align__(16) _Float16 p_lds[2][16 * 32];
    __shared__ float combine[16 * HDIM];

    int lane = threadIdx.x & 31;
    int c    = threadIdx.x >> 5;             // channel 0/1
    int qt = blockIdx.x, h = blockIdx.y;
    int half = lane >> 4, l16 = lane & 15;
    int qrow0 = qt * 16;

    const _Float16* Qh = Q  + h * DMH + c * HDIM;   // lda = QKN
    const _Float16* Kh = Kk + h * DMH + c * HDIM;
    const _Float16* Vh = VT + (size_t)h * HDIM * S_LEN;  // ldb = S_LEN

    // Q fragments for this 16-row tile (4 chunks of K=32 over HD=128)
    v16h qf[4];
#pragma unroll
    for (int d = 0; d < 4; ++d) qf[d] = load_a_frag(Qh, QKN, qrow0, d * 32, lane);

    v8f   acc[8] = {};                 // 16x128 output, 8 N-chunks
    float mst[8], lst[8];
#pragma unroll
    for (int r = 0; r < 8; ++r) { mst[r] = -1e30f; lst[r] = 0.f; }

    int nkt = (16 * qt + 15) / 32 + 1; // causal: keys [0, 16*qt+15]
    for (int kt = 0; kt < nkt; ++kt) {
        int key0 = kt * 32;

        if (kt + 1 < nkt) {
            // stream next K tile rows and next V tile columns toward cache
            __builtin_prefetch(Kh + (size_t)(key0 + 32 + lane) * QKN, 0, 1);
            __builtin_prefetch(Vh + (size_t)(lane * 4) * S_LEN + key0 + 32, 0, 1);
        }

        // ---- scores: S = Q * K^T over HD (two 16x16 frags for 32 keys) ----
        v8f s0 = {}, s1 = {};
#pragma unroll
        for (int d = 0; d < 4; ++d) {
            v16h kb0 = load_b_frag(Kh, QKN, key0,      d * 32, lane);
            v16h kb1 = load_b_frag(Kh, QKN, key0 + 16, d * 32, lane);
            s0 = wmma_f16(qf[d], kb0, s0);
            s1 = wmma_f16(qf[d], kb1, s1);
        }

        // ---- scale, causal mask, online softmax row stats ----
        float corr[8];
#pragma unroll
        for (int r = 0; r < 8; ++r) {
            int qrow = qrow0 + r + half * 8;
            float v0 = s0[r] * INV_SQRT_HD;
            float v1 = s1[r] * INV_SQRT_HD;
            if (key0 + l16      > qrow) v0 = -1e30f;
            if (key0 + 16 + l16 > qrow) v1 = -1e30f;
            float tm = fmaxf(v0, v1);
#pragma unroll
            for (int m = 1; m < 16; m <<= 1) tm = fmaxf(tm, __shfl_xor(tm, m, 32));
            float mnew = fmaxf(mst[r], tm);
            float cr = __expf(mst[r] - mnew);
            float p0 = __expf(v0 - mnew), p1 = __expf(v1 - mnew);
            float ps = p0 + p1;
#pragma unroll
            for (int m = 1; m < 16; m <<= 1) ps += __shfl_xor(ps, m, 32);
            lst[r] = lst[r] * cr + ps;
            mst[r] = mnew; corr[r] = cr;
            s0[r] = p0; s1[r] = p1;
        }
#pragma unroll
        for (int ch = 0; ch < 8; ++ch)
#pragma unroll
            for (int r = 0; r < 8; ++r) acc[ch][r] *= corr[r];

        // ---- relayout P (C-frag f32 -> A-frag f16) through LDS ----
#pragma unroll
        for (int r = 0; r < 8; ++r) {
            int row = r + half * 8;
            p_lds[c][row * 32 + l16]      = (_Float16)s0[r];
            p_lds[c][row * 32 + 16 + l16] = (_Float16)s1[r];
        }
        __syncthreads();
        v16h pf = load_a_frag(&p_lds[c][0], 32, 0, 0, lane);
        __syncthreads();

        // ---- acc += P (16x32) * V (32x128), 8 WMMAs ----
#pragma unroll
        for (int ch = 0; ch < 8; ++ch) {
            v16h vf = load_b_frag(Vh, S_LEN, ch * 16, key0, lane);
            acc[ch] = wmma_f16(pf, vf, acc[ch]);
        }
    }

    // ---- epilogue: combine channels, head RMSNorm, (1-lambda) scale ----
    float inv[8];
#pragma unroll
    for (int r = 0; r < 8; ++r) inv[r] = 1.0f / lst[r];

    if (c == 1) {
        float scl = scaleArr[h];
#pragma unroll
        for (int ch = 0; ch < 8; ++ch)
#pragma unroll
            for (int r = 0; r < 8; ++r)
                combine[(r + half * 8) * HDIM + ch * 16 + l16] = acc[ch][r] * inv[r] * scl;
    }
    __syncthreads();
    if (c == 0) {
#pragma unroll
        for (int ch = 0; ch < 8; ++ch)
#pragma unroll
            for (int r = 0; r < 8; ++r)
                acc[ch][r] = acc[ch][r] * inv[r] +
                             combine[(r + half * 8) * HDIM + ch * 16 + l16];
        float osc = 1.0f - lam[h];
#pragma unroll
        for (int r = 0; r < 8; ++r) {
            float ss = 0.f;
#pragma unroll
            for (int ch = 0; ch < 8; ++ch) ss += acc[ch][r] * acc[ch][r];
#pragma unroll
            for (int m = 1; m < 16; m <<= 1) ss += __shfl_xor(ss, m, 32);
            float rms = rsqrtf(ss / (float)HDIM + EPS_F);
            int row = qrow0 + r + half * 8;
#pragma unroll
            for (int ch = 0; ch < 8; ++ch) {
                int col = ch * 16 + l16;
                attOut[(size_t)row * D_MODEL + h * HDIM + col] =
                    (_Float16)(acc[ch][r] * rms * head_w[col] * osc);
            }
        }
    }
}

// ---------------------------------------------------------------------------
// launcher
// ---------------------------------------------------------------------------
extern "C" void kernel_launch(void* const* d_in, const int* in_sizes, int n_in,
                              void* d_out, int out_size, void* d_ws, size_t ws_size,
                              hipStream_t stream) {
    const float* x     = (const float*)d_in[0];
    const float* pre_w = (const float*)d_in[1];
    const float* wq    = (const float*)d_in[2];
    const float* wk    = (const float*)d_in[3];
    const float* wv    = (const float*)d_in[4];
    const float* wo    = (const float*)d_in[5];
    const float* hnw   = (const float*)d_in[6];
    const float* q1    = (const float*)d_in[7];
    const float* q2    = (const float*)d_in[8];
    const float* k1    = (const float*)d_in[9];
    const float* k2    = (const float*)d_in[10];
    const float* lam   = (const float*)d_in[11];
    float* out = (float*)d_out;

    char* w = (char*)d_ws;
    _Float16* h16   = (_Float16*)(w + 0);           //  8 MB  S*D
    _Float16* wq16  = (_Float16*)(w + 8388608);     // 16 MB  4096*2048
    _Float16* wk16  = (_Float16*)(w + 25165824);    // 16 MB
    _Float16* wv16  = (_Float16*)(w + 41943040);    //  8 MB
    _Float16* wo16  = (_Float16*)(w + 50331648);    //  8 MB
    _Float16* q16   = (_Float16*)(w + 58720256);    // 16 MB  S*4096
    _Float16* k16   = (_Float16*)(w + 75497472);    // 16 MB
    _Float16* v16   = (_Float16*)(w + 92274688);    //  8 MB  S*2048
    _Float16* vT16  = (_Float16*)(w + 100663296);   //  8 MB
    _Float16* att16 = (_Float16*)(w + 109051904);   //  8 MB
    float*    scl   = (float*)   (w + 117440512);   // 64 B

    // 1) pre-norm
    rmsnorm_kernel<<<S_LEN, 256, 0, stream>>>(x, pre_w, h16, D_MODEL);

    // 2) weight conversion
    cvt_f32_f16_kernel<<<(QKN*D_MODEL)/256, 256, 0, stream>>>(wq, wq16, (size_t)QKN*D_MODEL);
    cvt_f32_f16_kernel<<<(QKN*D_MODEL)/256, 256, 0, stream>>>(wk, wk16, (size_t)QKN*D_MODEL);
    cvt_f32_f16_kernel<<<(D_MODEL*D_MODEL)/256, 256, 0, stream>>>(wv, wv16, (size_t)D_MODEL*D_MODEL);
    cvt_f32_f16_kernel<<<(D_MODEL*D_MODEL)/256, 256, 0, stream>>>(wo, wo16, (size_t)D_MODEL*D_MODEL);

    // 3) QKV projections (WMMA)
    gemm_f16_wmma_kernel<<<dim3(S_LEN/128, QKN/128), 256, 0, stream>>>(
        h16, wq16, S_LEN, QKN, D_MODEL, q16, nullptr, nullptr);
    gemm_f16_wmma_kernel<<<dim3(S_LEN/128, QKN/128), 256, 0, stream>>>(
        h16, wk16, S_LEN, QKN, D_MODEL, k16, nullptr, nullptr);
    gemm_f16_wmma_kernel<<<dim3(S_LEN/128, D_MODEL/128), 256, 0, stream>>>(
        h16, wv16, S_LEN, D_MODEL, D_MODEL, v16, nullptr, nullptr);

    // 4) RoPE on q and k
    rope_kernel<<<(S_LEN*NHEADS*DMH/2 + 255)/256, 256, 0, stream>>>(q16, NHEADS, DMH);
    rope_kernel<<<(S_LEN*NHEADS*DMH/2 + 255)/256, 256, 0, stream>>>(k16, NHEADS, DMH);

    // 5) V transpose for contiguous P*V B-fragments
    vtrans_kernel<<<(S_LEN*D_MODEL)/256, 256, 0, stream>>>(v16, vT16);

    // 6) differential scale
    scale_kernel<<<1, 32, 0, stream>>>(q1, q2, k1, k2, lam, scl);

    // 7) flash differential attention
    flashdiff_kernel<<<dim3(S_LEN/16, NHEADS), 64, 0, stream>>>(
        q16, k16, vT16, scl, hnw, lam, att16);

    // 8) output projection + residual (WMMA, f32 epilogue)
    gemm_f16_wmma_kernel<<<dim3(S_LEN/128, D_MODEL/128), 256, 0, stream>>>(
        att16, wo16, S_LEN, D_MODEL, D_MODEL, nullptr, out, x);
}
